// AttentionHead_89687507076307
// MI455X (gfx1250) — compile-verified, matrix operand order checked by reading, not run on hardware
//
#include <hip/hip_runtime.h>
#include <hip/hip_bf16.h>

typedef __attribute__((ext_vector_type(16))) _Float16 v16h;
typedef __attribute__((ext_vector_type(8)))  _Float16 v8h;
typedef __attribute__((ext_vector_type(8)))  float    v8f;

#define WMMA_F16(a, b, c) \
    __builtin_amdgcn_wmma_f32_16x16x32_f16(false, (a), false, (b), (short)0, (c), false, false)

// ---------------------------------------------------------------------------
// Fragment loaders (CDNA5 wave32 WMMA layouts, CDNA5 ISA §7.12.2)
// A 16x32 f16: lanes 0-15 row M=lane, K = {0..7, 16..23};
//              lanes 16-31 row M=lane-16, K = {8..15, 24..31}
// The f16 loaders read two contiguous 16-byte runs per lane (f32 loader: 32-byte runs) -> b128 loads.
// ---------------------------------------------------------------------------
__device__ inline v16h load_a_f16(const _Float16* src, int ld) {
    const int lane = threadIdx.x & 31;
    const int r    = lane & 15;
    const int kb   = (lane >> 4) << 3;           // 0 or 8
    const _Float16* p = src + r * ld + kb;
    const v8h lo = *(const v8h*)(p);
    const v8h hi = *(const v8h*)(p + 16);
    v16h a;
#pragma unroll
    for (int i = 0; i < 8; ++i) { a[i] = lo[i]; a[8 + i] = hi[i]; }
    return a;
}

__device__ inline v16h load_a_f32cvt(const float* src, int ld) {
    const int lane = threadIdx.x & 31;
    const int r    = lane & 15;
    const int kb   = (lane >> 4) << 3;
    const float* p = src + r * ld + kb;
    v16h a;
#pragma unroll
    for (int i = 0; i < 8; ++i) a[i]     = (_Float16)p[i];
#pragma unroll
    for (int i = 0; i < 8; ++i) a[8 + i] = (_Float16)p[16 + i];
    return a;
}

// B 32x16 f16 where each column n is contiguous in memory (k advances by 1
// element, columns are col_stride apart). Covers W^T, K^T and V^T layouts.
__device__ inline v16h load_b_colmajor(const _Float16* src, int col_stride) {
    const int lane = threadIdx.x & 31;
    const int n    = lane & 15;
    const int kb   = (lane >> 4) << 3;
    const _Float16* p = src + (size_t)n * col_stride + kb;
    const v8h lo = *(const v8h*)(p);
    const v8h hi = *(const v8h*)(p + 16);
    v16h b;
#pragma unroll
    for (int i = 0; i < 8; ++i) { b[i] = lo[i]; b[8 + i] = hi[i]; }
    return b;
}

// ---------------------------------------------------------------------------
// Kernel 1: convert the three 1024x64 fp32 weights to f16 *transposed*
// (W^T [64][1024]) so projection B-fragments are column-contiguous.
// ---------------------------------------------------------------------------
__global__ __launch_bounds__(256) void cvt_weights_T(const float* __restrict__ kw,
                                                     const float* __restrict__ vw,
                                                     const float* __restrict__ qw,
                                                     _Float16* __restrict__ khT,
                                                     _Float16* __restrict__ vhT,
                                                     _Float16* __restrict__ qhT,
                                                     int n) {
    int i = blockIdx.x * blockDim.x + threadIdx.x;
    if (i < n) {
        const int k = i >> 6;          // 0..1023
        const int c = i & 63;          // 0..63
        const int o = c * 1024 + k;
        khT[o] = (_Float16)kw[i];
        vhT[o] = (_Float16)vw[i];
        qhT[o] = (_Float16)qw[i];
    }
}

// ---------------------------------------------------------------------------
// Kernel 2: projection Y = X[M,1024] @ W[1024,64], f16 WMMA, f32 acc.
// One wave computes a 16-row x 64-col tile; 32 k-steps of 16x16x32.
// transposed==0: Y[row][col] row-major.  transposed==1: Y^T[b][col][s]
// (per-batch transpose, S=2048) for the values tensor.
// Output staged through LDS -> 4 coalesced global_store_b128 per lane.
// ---------------------------------------------------------------------------
__global__ __launch_bounds__(32) void proj16(const float* __restrict__ X,
                                             const _Float16* __restrict__ WhT,
                                             _Float16* __restrict__ Y,
                                             int transposed) {
    const int D_IN = 1024, S = 2048;
    const int rbase = blockIdx.x * 16;
    v8f acc0 = {}, acc1 = {}, acc2 = {}, acc3 = {};

    for (int k = 0; k < D_IN; k += 32) {
        v16h a = load_a_f32cvt(X + (size_t)rbase * D_IN + k, D_IN);
        acc0 = WMMA_F16(a, load_b_colmajor(WhT +  0 * 1024 + k, 1024), acc0);
        acc1 = WMMA_F16(a, load_b_colmajor(WhT + 16 * 1024 + k, 1024), acc1);
        acc2 = WMMA_F16(a, load_b_colmajor(WhT + 32 * 1024 + k, 1024), acc2);
        acc3 = WMMA_F16(a, load_b_colmajor(WhT + 48 * 1024 + k, 1024), acc3);
    }

    // C/D layout: VGPR v, lanes 0-15 -> (row v, col lane); lanes 16-31 -> row v+8
    const int lane = threadIdx.x & 31;
    const int hi   = lane >> 4;
    const int c0   = lane & 15;
    __shared__ _Float16 tile[16 * 64];   // normal: [row][col]; transposed: [col][row]

    if (!transposed) {
#pragma unroll
        for (int v = 0; v < 8; ++v) {
            const int r = v + 8 * hi;
            tile[r * 64 +  0 + c0] = (_Float16)acc0[v];
            tile[r * 64 + 16 + c0] = (_Float16)acc1[v];
            tile[r * 64 + 32 + c0] = (_Float16)acc2[v];
            tile[r * 64 + 48 + c0] = (_Float16)acc3[v];
        }
        __syncthreads();
#pragma unroll
        for (int j = 0; j < 4; ++j) {
            const int c   = lane + 32 * j;       // 128 x 16B chunks
            const int r   = c >> 3;
            const int off = (c & 7) * 8;
            *(v8h*)(Y + (size_t)(rbase + r) * 64 + off) = *(const v8h*)(tile + r * 64 + off);
        }
    } else {
#pragma unroll
        for (int v = 0; v < 8; ++v) {
            const int r = v + 8 * hi;            // local s index 0..15
            tile[( 0 + c0) * 16 + r] = (_Float16)acc0[v];
            tile[(16 + c0) * 16 + r] = (_Float16)acc1[v];
            tile[(32 + c0) * 16 + r] = (_Float16)acc2[v];
            tile[(48 + c0) * 16 + r] = (_Float16)acc3[v];
        }
        __syncthreads();
        const int b  = rbase >> 11;              // rbase / S
        const int s0 = rbase & (S - 1);
#pragma unroll
        for (int j = 0; j < 4; ++j) {
            const int c   = lane + 32 * j;       // 128 x 16B chunks (64 e-rows x 2)
            const int e   = c >> 1;
            const int off = (c & 1) * 8;
            *(v8h*)(Y + ((size_t)b * 64 + e) * S + s0 + off) =
                *(const v8h*)(tile + e * 16 + off);
        }
    }
}

// ---------------------------------------------------------------------------
// Kernel 3: causal flash attention. One wave owns 16 query rows of one batch.
// KV tiles of 32 keys; online softmax; all matmuls f16 WMMA w/ f32 acc.
// K row-major [b][s][64] (K^T columns contiguous); V transposed [b][64][s].
// ---------------------------------------------------------------------------
__global__ __launch_bounds__(32) void flash16(const _Float16* __restrict__ Qh,
                                              const _Float16* __restrict__ Kh,
                                              const _Float16* __restrict__ VhT,
                                              float* __restrict__ O) {
    const int S = 2048, E = 64;
    const int qb = blockIdx.x * 16;
    const int b  = blockIdx.y;

    const int lane = threadIdx.x & 31;
    const int hi   = lane >> 4;
    const int c0   = lane & 15;

    // Q tile in A layout: 16 rows x 64 -> two 16x32 fragments
    const _Float16* qp = Qh + (size_t)(b * S + qb) * E;
    const v16h aq0 = load_a_f16(qp,      E);
    const v16h aq1 = load_a_f16(qp + 32, E);

    v8f o0 = {}, o1 = {}, o2 = {}, o3 = {};
    float m[8], l[8];
#pragma unroll
    for (int v = 0; v < 8; ++v) { m[v] = -1e30f; l[v] = 0.0f; }

    __shared__ _Float16 pshare[16 * 32];

    for (int jb = 0; jb < qb + 16; jb += 32) {
        const _Float16* kp = Kh  + (size_t)(b * S + jb) * E;
        const _Float16* vp = VhT + (size_t)b * 64 * S + jb;   // V^T: [e][s]

        // S = Q K^T / sqrt(64): two 16x16 fragments (keys jb..+15, jb+16..+31)
        v8f s0 = {}, s1 = {};
        s0 = WMMA_F16(aq0, load_b_colmajor(kp,               E), s0);  // e 0..31
        s0 = WMMA_F16(aq1, load_b_colmajor(kp + 32,          E), s0);  // e 32..63
        s1 = WMMA_F16(aq0, load_b_colmajor(kp + 16 * E,      E), s1);
        s1 = WMMA_F16(aq1, load_b_colmajor(kp + 16 * E + 32, E), s1);

        const bool need_mask = (jb + 31 > qb);
#pragma unroll
        for (int v = 0; v < 8; ++v) {
            s0[v] *= 0.125f;
            s1[v] *= 0.125f;
            if (need_mask) {
                const int row = qb + v + 8 * hi;
                if (jb + c0      > row) s0[v] = -1e30f;
                if (jb + 16 + c0 > row) s1[v] = -1e30f;
            }
        }

        // Online softmax: per-row stats via half-wave (width-16) reductions.
#pragma unroll
        for (int v = 0; v < 8; ++v) {
            float rmax = fmaxf(s0[v], s1[v]);
#pragma unroll
            for (int msk = 8; msk >= 1; msk >>= 1)
                rmax = fmaxf(rmax, __shfl_xor(rmax, msk, 16));
            const float mnew = fmaxf(m[v], rmax);
            const float p0 = __expf(s0[v] - mnew);
            const float p1 = __expf(s1[v] - mnew);
            pshare[(v + 8 * hi) * 32 + c0]      = (_Float16)p0;
            pshare[(v + 8 * hi) * 32 + 16 + c0] = (_Float16)p1;
            float rs = p0 + p1;
#pragma unroll
            for (int msk = 8; msk >= 1; msk >>= 1)
                rs += __shfl_xor(rs, msk, 16);
            const float alpha = __expf(m[v] - mnew);
            l[v] = l[v] * alpha + rs;
            m[v] = mnew;
            o0[v] *= alpha; o1[v] *= alpha; o2[v] *= alpha; o3[v] *= alpha;
        }
        __syncthreads();   // LDS writes visible before re-staged A load

        // P (16x32) back as A fragment; O += P @ V (four 16-col output tiles)
        const v16h pa = load_a_f16(pshare, 32);
        o0 = WMMA_F16(pa, load_b_colmajor(vp +  0 * S, S), o0);
        o1 = WMMA_F16(pa, load_b_colmajor(vp + 16 * S, S), o1);
        o2 = WMMA_F16(pa, load_b_colmajor(vp + 32 * S, S), o2);
        o3 = WMMA_F16(pa, load_b_colmajor(vp + 48 * S, S), o3);
        __syncthreads();   // protect pshare before next iteration overwrites
    }

    // Final normalize and store fp32 output (coalesced across lanes).
#pragma unroll
    for (int v = 0; v < 8; ++v) {
        const float inv = 1.0f / l[v];
        const size_t row = (size_t)(b * S + qb + v + 8 * hi) * E;
        O[row +  0 + c0] = o0[v] * inv;
        O[row + 16 + c0] = o1[v] * inv;
        O[row + 32 + c0] = o2[v] * inv;
        O[row + 48 + c0] = o3[v] * inv;
    }
}

// ---------------------------------------------------------------------------
// Host-side launch
// ---------------------------------------------------------------------------
extern "C" void kernel_launch(void* const* d_in, const int* in_sizes, int n_in,
                              void* d_out, int out_size, void* d_ws, size_t ws_size,
                              hipStream_t stream) {
    const int B = 8, S = 2048, D_IN = 1024, D_OUT = 64;
    const int M = B * S;                 // 16384 rows
    const int WN = D_IN * D_OUT;         // 65536 weight elements

    const float* xk = (const float*)d_in[0];
    const float* xv = (const float*)d_in[1];
    const float* xq = (const float*)d_in[2];
    const float* kw = (const float*)d_in[3];
    const float* vw = (const float*)d_in[4];
    const float* qw = (const float*)d_in[5];
    float* out = (float*)d_out;

    // Workspace (f16): 3 transposed weights (128KB each) + keys (2MB, row-major)
    // + values^T (2MB, [b][64][S]) + queries (2MB, row-major)
    char* ws = (char*)d_ws;
    _Float16* khT = (_Float16*)(ws);
    _Float16* vhT = (_Float16*)(ws + 131072);
    _Float16* qhT = (_Float16*)(ws + 262144);
    _Float16* keys    = (_Float16*)(ws + 393216);
    _Float16* valuesT = (_Float16*)(ws + 393216 + 2097152);
    _Float16* queries = (_Float16*)(ws + 393216 + 2 * 2097152);

    cvt_weights_T<<<(WN + 255) / 256, 256, 0, stream>>>(kw, vw, qw, khT, vhT, qhT, WN);

    proj16<<<M / 16, 32, 0, stream>>>(xk, khT, keys,    0);
    proj16<<<M / 16, 32, 0, stream>>>(xv, vhT, valuesT, 1);
    proj16<<<M / 16, 32, 0, stream>>>(xq, qhT, queries, 0);

    dim3 grid(S / 16, B);
    flash16<<<grid, 32, 0, stream>>>(queries, keys, valuesT, out);
}
